// MultiheadAttention_57638461112519
// MI455X (gfx1250) — matrix operand order repaired and checked by reading, not compiled, not run on hardware
//
#include <hip/hip_runtime.h>
#include <hip/hip_bf16.h>

// ---------------------------------------------------------------------------
// MHA forward for MI455X (gfx1250, wave32, WMMA bf16, async global->LDS).
// B=4, T=2048, DIM=1024, H=16, HD=64.
// ---------------------------------------------------------------------------

typedef __attribute__((ext_vector_type(16))) __bf16 v16bf;
typedef __attribute__((ext_vector_type(8)))  __bf16 v8bf;
typedef __attribute__((ext_vector_type(4)))  __bf16 v4bf;
typedef __attribute__((ext_vector_type(8)))  float  v8f;
typedef __attribute__((ext_vector_type(4)))  float  v4f;
typedef __attribute__((ext_vector_type(4)))  int    v4i;

union FragBF { v16bf v; v8bf h[2]; };

__device__ __forceinline__ v8f wmma_bf16(v16bf a, v16bf b, v8f c) {
  return __builtin_amdgcn_wmma_f32_16x16x32_bf16(false, a, false, b,
                                                 (short)0, c, false, false);
}

#define BATCH 4
#define TLEN  2048
#define DIM   1024
#define NHEAD 16
#define HDIM  64
#define MROWS (BATCH * TLEN)                          // 8192
#define BHSZ  ((size_t)BATCH * NHEAD * TLEN * HDIM)   // elems per Q/K/V buffer

// ---- gfx1250 async global->LDS path (guarded; falls back to plain copy) ----
#if defined(__has_builtin)
#if __has_builtin(__builtin_amdgcn_global_load_async_to_lds_b128) && \
    __has_builtin(__builtin_amdgcn_s_wait_asynccnt)
#define USE_ASYNC 1
#endif
#endif
#ifndef USE_ASYNC
#define USE_ASYNC 0
#endif

typedef __attribute__((address_space(1))) v4i gv4i_t;  // global int4
typedef __attribute__((address_space(3))) v4i lv4i_t;  // LDS int4

__device__ __forceinline__ void copy16_g2l(const __bf16* g, __bf16* l) {
#if USE_ASYNC
  gv4i_t* gp = (gv4i_t*)(unsigned long long)(const void*)g;
  lv4i_t* lp = (lv4i_t*)(unsigned long long)(void*)l;
  __builtin_amdgcn_global_load_async_to_lds_b128(gp, lp, 0, 0);
#else
  *(v8bf*)l = *(const v8bf*)g;
#endif
}

__device__ __forceinline__ void wait_async() {
#if USE_ASYNC
  __builtin_amdgcn_s_wait_asynccnt(0);
#endif
}

// ---------------------------------------------------------------------------
// Kernel 0: f32 -> bf16 bulk conversion (for X so GEMM A-tiles can go async).
// ---------------------------------------------------------------------------
__global__ __launch_bounds__(256) void k_cvt_bf16(const float* __restrict__ in,
                                                  __bf16* __restrict__ out) {
  int i = (blockIdx.x * 256 + threadIdx.x) * 4;
  v4f f = *(const v4f*)(in + i);
  v4bf b;
#pragma unroll
  for (int j = 0; j < 4; j++) b[j] = (__bf16)f[j];
  *(v4bf*)(out + i) = b;
}

// ---------------------------------------------------------------------------
// Unified GEMM: C[128x128 tile] = A(bf16) @ W(f32->bf16) ; 8 waves, K-step 32.
// EPI 0: QKV epilogue (bias + scatter into Q[b,h,t,d], K[b,h,t,d], Vt[b,h,d,t])
// EPI 1: f32 output epilogue (bias + store to Out)
// ---------------------------------------------------------------------------
template <int N, int EPI>
__global__ __launch_bounds__(256) void k_gemm(
    const __bf16* __restrict__ Ain, const float* __restrict__ W,
    const float* __restrict__ bias, float* __restrict__ OutF,
    __bf16* __restrict__ Qh, __bf16* __restrict__ Kh, __bf16* __restrict__ Vt)
{
  constexpr int K = 1024;
  __shared__ __align__(16) __bf16 As[128 * 40];   // [m][k] padded
  __shared__ __align__(16) __bf16 Bs[128 * 40];   // [n][k] padded (transposed)

  const int tid  = threadIdx.x;
  const int wid  = tid >> 5, lane = tid & 31;
  const int wm   = wid & 3,  wn   = wid >> 2;     // wave grid 4(m) x 2(n)
  const int half = lane >> 4, l16 = lane & 15;
  const int m0   = blockIdx.y * 128, n0 = blockIdx.x * 128;

  const int ar  = tid >> 1,        ac  = (tid & 1) * 16;  // A: 128r x 32k
  const int br2 = (tid >> 4) * 2;                          // B: k row pair
  const int bc2 = (tid & 15) * 8;                          // B: 8 columns

  v8f acc[2][4] = {};

  for (int k0 = 0; k0 < K; k0 += 32) {
    const __bf16* srcA  = Ain + (size_t)(m0 + ar) * K + k0 + ac;
    const float*  srcB0 = W + (size_t)(k0 + br2) * N + n0 + bc2;
    const float*  srcB1 = srcB0 + N;
    if (k0 + 32 < K) {
      __builtin_prefetch(srcA + 32, 0, 3);
      __builtin_prefetch(srcB0 + (size_t)32 * N, 0, 3);
    }
    // stage A: bf16 rows, async global->LDS
    copy16_g2l(srcA,     &As[ar * 40 + ac]);
    copy16_g2l(srcA + 8, &As[ar * 40 + ac + 8]);
    // stage B transposed: pack two k-rows into one b32 per column
    {
      v4f f0a = *(const v4f*)(srcB0);
      v4f f0b = *(const v4f*)(srcB0 + 4);
      v4f f1a = *(const v4f*)(srcB1);
      v4f f1b = *(const v4f*)(srcB1 + 4);
#pragma unroll
      for (int e = 0; e < 8; e++) {
        float lo = (e < 4) ? f0a[e] : f0b[e - 4];
        float hi = (e < 4) ? f1a[e] : f1b[e - 4];
        union { __bf16 b[2]; unsigned u; } pk;
        pk.b[0] = (__bf16)lo;  // k = br2
        pk.b[1] = (__bf16)hi;  // k = br2+1
        *(unsigned*)&Bs[(bc2 + e) * 40 + br2] = pk.u;
      }
    }
    wait_async();
    __syncthreads();

    FragBF af[2];
#pragma unroll
    for (int mt = 0; mt < 2; mt++) {
      int row = wm * 32 + mt * 16 + l16;
      af[mt].h[0] = *(const v8bf*)&As[row * 40 + 8 * half];
      af[mt].h[1] = *(const v8bf*)&As[row * 40 + 16 + 8 * half];
    }
#pragma unroll
    for (int nt = 0; nt < 4; nt++) {
      FragBF bfr;
      int col = wn * 64 + nt * 16 + l16;
      bfr.h[0] = *(const v8bf*)&Bs[col * 40 + 8 * half];
      bfr.h[1] = *(const v8bf*)&Bs[col * 40 + 16 + 8 * half];
#pragma unroll
      for (int mt = 0; mt < 2; mt++)
        acc[mt][nt] = wmma_bf16(af[mt].v, bfr.v, acc[mt][nt]);
    }
    __syncthreads();
  }

#pragma unroll
  for (int nt = 0; nt < 4; nt++) {
    int n = n0 + wn * 64 + nt * 16 + l16;
    float bv = bias[n];
    if constexpr (EPI == 0) {
      int sect = n >> 10;              // 0=Q 1=K 2=V (uniform across wave)
      int f = n & 1023, h = f >> 6, d = f & 63;
#pragma unroll
      for (int mt = 0; mt < 2; mt++) {
        int mbase = m0 + wm * 32 + mt * 16 + 8 * half;
#pragma unroll
        for (int i = 0; i < 8; i++) {
          int m = mbase + i;
          int b = m >> 11, t = m & 2047;
          float v = acc[mt][nt][i] + bv;
          size_t bh = (size_t)(b * NHEAD + h);
          if (sect == 0)      Qh[(bh * TLEN + t) * HDIM + d] = (__bf16)v;
          else if (sect == 1) Kh[(bh * TLEN + t) * HDIM + d] = (__bf16)v;
          else                Vt[(bh * HDIM + d) * TLEN + t] = (__bf16)v;
        }
      }
    } else {
#pragma unroll
      for (int mt = 0; mt < 2; mt++) {
        int mbase = m0 + wm * 32 + mt * 16 + 8 * half;
#pragma unroll
        for (int i = 0; i < 8; i++)
          OutF[(size_t)(mbase + i) * N + n] = acc[mt][nt][i] + bv;
      }
    }
  }
}

// ---------------------------------------------------------------------------
// Kernel 2: interleaved RoPE on Q and K, in place. One thread per (b,h,t,pair).
// ---------------------------------------------------------------------------
__global__ __launch_bounds__(256) void k_rope(__bf16* __restrict__ Qh,
                                              __bf16* __restrict__ Kh)
{
  int idx = blockIdx.x * 256 + threadIdx.x;   // BATCH*NHEAD*TLEN*32 total
  int p  = idx & 31;
  int t  = (idx >> 5) & (TLEN - 1);
  int bh = idx >> 16;                         // TLEN*32 == 1<<16
  float inv = __expf(-9.2103403719761836f * (float)(2 * p) * (1.0f / 64.0f));
  float ang = (float)t * inv;
  float s, c;
  __sincosf(ang, &s, &c);
  size_t base = ((size_t)bh * TLEN + t) * HDIM + 2 * p;
  float q0 = (float)Qh[base], q1 = (float)Qh[base + 1];
  Qh[base]     = (__bf16)(q0 * c - q1 * s);
  Qh[base + 1] = (__bf16)(q1 * c + q0 * s);
  float k0 = (float)Kh[base], k1 = (float)Kh[base + 1];
  Kh[base]     = (__bf16)(k0 * c - k1 * s);
  Kh[base + 1] = (__bf16)(k1 * c + k0 * s);
}

// ---------------------------------------------------------------------------
// Kernel 3: causal flash attention. One wave per (b, h, 32-query block).
// K/V tiles async-staged into LDS per key block; S = QK^T/8 via WMMA; online
// softmax via shfl over 16-lane halves; P -> LDS -> A-frags; O += P V (WMMA).
// ---------------------------------------------------------------------------
__global__ __launch_bounds__(32) void k_attn(
    const __bf16* __restrict__ Qh, const __bf16* __restrict__ Kh,
    const __bf16* __restrict__ Vt, __bf16* __restrict__ Outh)
{
  __shared__ __align__(16) __bf16 Ks[64 * 72];  // [key][d]
  __shared__ __align__(16) __bf16 Vs[64 * 72];  // [d][key]
  __shared__ __align__(16) __bf16 Pt[16 * 72];

  const int lane = threadIdx.x;
  const int half = lane >> 4, l16 = lane & 15;
  const int qb = blockIdx.x & 63;
  const int h  = (blockIdx.x >> 6) & 15;
  const int b  = blockIdx.x >> 10;
  const int qbase = qb * 32;
  const size_t bh = (size_t)(b * NHEAD + h);

  const __bf16* Qp = Qh + bh * TLEN * HDIM;   // [t][d]
  const __bf16* Kp = Kh + bh * TLEN * HDIM;   // [t][d]
  const __bf16* Vp = Vt + bh * HDIM * TLEN;   // [d][t]

  // Q fragments (A-layout): [mt][kstep over d]
  FragBF qf[2][2];
#pragma unroll
  for (int mt = 0; mt < 2; mt++)
#pragma unroll
    for (int ks = 0; ks < 2; ks++) {
      int row = qbase + mt * 16 + l16;
      qf[mt][ks].h[0] = *(const v8bf*)&Qp[row * HDIM + ks * 32 + 8 * half];
      qf[mt][ks].h[1] = *(const v8bf*)&Qp[row * HDIM + ks * 32 + 16 + 8 * half];
    }

  float mrow[2][8], lrow[2][8];
  v8f o[2][4];
  const v8f vzero = {};
#pragma unroll
  for (int mt = 0; mt < 2; mt++) {
#pragma unroll
    for (int i = 0; i < 8; i++) { mrow[mt][i] = -1e30f; lrow[mt][i] = 0.f; }
#pragma unroll
    for (int dt = 0; dt < 4; dt++) o[mt][dt] = vzero;
  }

  const int nkb = (qbase + 95) >> 6;          // causal bound on 64-key blocks
  for (int kb = 0; kb < nkb; kb++) {
    const int kbase = kb * 64;
    // ---- async-stage K (64 keys x d 0..31) and V (64 d x first 32 keys) into LDS ----
#pragma unroll
    for (int c = 0; c < 8; c++) {
      int chunk = c * 32 + lane;              // 0..255
      int r  = chunk >> 2;                    // 0..63
      int of = (chunk & 3) * 8;               // 0,8,16,24
      copy16_g2l(&Kp[(size_t)(kbase + r) * HDIM + of], &Ks[r * 72 + of]);
      copy16_g2l(&Vp[(size_t)r * TLEN + kbase + of],   &Vs[r * 72 + of]);
    }
    wait_async();   // single-wave workgroup: DS in-order, no barrier needed

#pragma unroll
    for (int mt = 0; mt < 2; mt++) {
      // ---- scores S[16 x 64] ----
      v8f s[4];
#pragma unroll
      for (int nt = 0; nt < 4; nt++) {
        int krow = nt * 16 + l16;
        FragBF kf;
        v8f z = vzero;
        kf.h[0] = *(const v8bf*)&Ks[krow * 72 + 8 * half];
        kf.h[1] = *(const v8bf*)&Ks[krow * 72 + 16 + 8 * half];
        z = wmma_bf16(qf[mt][0].v, kf.v, z);
        kf.h[0] = *(const v8bf*)&Ks[krow * 72 + 32 + 8 * half];
        kf.h[1] = *(const v8bf*)&Ks[krow * 72 + 32 + 16 + 8 * half];
        z = wmma_bf16(qf[mt][1].v, kf.v, z);
        s[nt] = z;
      }
      // ---- scale + causal mask ----
#pragma unroll
      for (int nt = 0; nt < 4; nt++) {
        int kcol = kbase + nt * 16 + l16;
#pragma unroll
        for (int i = 0; i < 8; i++) {
          int q = qbase + mt * 16 + 8 * half + i;
          float v = s[nt][i] * 0.125f;        // 1/sqrt(64)
          s[nt][i] = (kcol <= q) ? v : -1e30f;
        }
      }
      // ---- online softmax per row ----
#pragma unroll
      for (int i = 0; i < 8; i++) {
        float rm = fmaxf(fmaxf(s[0][i], s[1][i]), fmaxf(s[2][i], s[3][i]));
#pragma unroll
        for (int off = 1; off < 16; off <<= 1) rm = fmaxf(rm, __shfl_xor(rm, off));
        float mold = mrow[mt][i];
        float mnew = fmaxf(mold, rm);
        float alpha = __expf(mold - mnew);
        float ps = 0.f;
#pragma unroll
        for (int nt = 0; nt < 4; nt++) {
          float e = __expf(s[nt][i] - mnew);
          s[nt][i] = e;
          ps += e;
        }
#pragma unroll
        for (int off = 1; off < 16; off <<= 1) ps += __shfl_xor(ps, off);
        lrow[mt][i] = lrow[mt][i] * alpha + ps;
        mrow[mt][i] = mnew;
#pragma unroll
        for (int dt = 0; dt < 4; dt++) o[mt][dt][i] *= alpha;
      }
      // ---- P (C-layout) -> LDS -> A-layout fragments ----
#pragma unroll
      for (int nt = 0; nt < 4; nt++)
#pragma unroll
        for (int i = 0; i < 8; i++)
          Pt[(8 * half + i) * 72 + nt * 16 + l16] = (__bf16)s[nt][i];
      FragBF pf[2];
#pragma unroll
      for (int ks = 0; ks < 2; ks++) {
        pf[ks].h[0] = *(const v8bf*)&Pt[l16 * 72 + ks * 32 + 8 * half];
        pf[ks].h[1] = *(const v8bf*)&Pt[l16 * 72 + ks * 32 + 16 + 8 * half];
      }
      // ---- O += P @ V ----
#pragma unroll
      for (int dt = 0; dt < 4; dt++) {
#pragma unroll
        for (int ks = 0; ks < 2; ks++) {
          FragBF vf;
          const __bf16* vp = &Vs[(dt * 16 + l16) * 72 + ks * 32 + 8 * half];
          vf.h[0] = *(const v8bf*)vp;
          vf.h[1] = *(const v8bf*)(vp + 16);
          o[mt][dt] = wmma_bf16(pf[ks].v, vf.v, o[mt][dt]);
        }
      }
    } // mt
  } // kb

  // ---- normalize and write [b,t, h*64+d] as bf16 ----
#pragma unroll
  for (int mt = 0; mt < 2; mt++)
#pragma unroll
    for (int dt = 0; dt < 4; dt++)
#pragma unroll
      for (int i = 0; i < 8; i++) {
        int q = qbase + mt * 16 + 8 * half + i;
        int d = dt * 16 + l16;
        float val = o[mt][dt][i] / lrow[mt][i];
        Outh[((size_t)b * TLEN + q) * DIM + h * HDIM + d] = (__bf16)val;
      }
}

// ---------------------------------------------------------------------------
extern "C" void kernel_launch(void* const* d_in, const int* in_sizes, int n_in,
                              void* d_out, int out_size, void* d_ws, size_t ws_size,
                              hipStream_t stream) {
  (void)in_sizes; (void)n_in; (void)out_size; (void)ws_size;
  const float* x      = (const float*)d_in[0];
  // d_in[1] is the causal mask; causality computed analytically.
  const float* qkv_w  = (const float*)d_in[2];
  const float* qkv_b  = (const float*)d_in[3];
  const float* out_w  = (const float*)d_in[4];
  const float* out_b  = (const float*)d_in[5];
  float* out = (float*)d_out;

  __bf16* Qh   = (__bf16*)d_ws;                    // [b,h,t,d]  16 MB
  __bf16* Kh   = Qh + BHSZ;                        // [b,h,t,d]  16 MB
  __bf16* Vt   = Kh + BHSZ;                        // [b,h,d,t]  16 MB
  __bf16* Attn = Vt + BHSZ;                        // [b*t, dim] 16 MB
  __bf16* Xb   = Attn + (size_t)MROWS * DIM;       // [b*t, dim] 16 MB

  // 0) x -> bf16 (enables async A-tile staging and halves X re-read traffic)
  k_cvt_bf16<<<(MROWS * DIM / 4) / 256, 256, 0, stream>>>(x, Xb);
  // 1) QKV projection + bias + head scatter (V transposed)
  k_gemm<3 * DIM, 0><<<dim3(3 * DIM / 128, MROWS / 128), 256, 0, stream>>>(
      Xb, qkv_w, qkv_b, nullptr, Qh, Kh, Vt);
  // 2) RoPE on Q and K
  k_rope<<<(BATCH * NHEAD * TLEN * 32) / 256, 256, 0, stream>>>(Qh, Kh);
  // 3) causal flash attention
  k_attn<<<BATCH * NHEAD * (TLEN / 32), 32, 0, stream>>>(Qh, Kh, Vt, Attn);
  // 4) output projection
  k_gemm<DIM, 1><<<dim3(DIM / 128, MROWS / 128), 256, 0, stream>>>(
      Attn, out_w, out_b, out, nullptr, nullptr, nullptr);
}
